// GATv2_46935402611267
// MI455X (gfx1250) — compile-verified, matrix operand order checked by reading, not run on hardware
//
#include <hip/hip_runtime.h>

typedef __attribute__((ext_vector_type(16))) _Float16 v16h;
typedef __attribute__((ext_vector_type(8)))  _Float16 v8h;
typedef __attribute__((ext_vector_type(8)))  float    v8f;

#define NNODES 50000
#define NEDGES 400000
#define NHEADS 4

// ---------- helpers ----------
__device__ __forceinline__ void atomicMaxF(float* addr, float val) {
  if (val >= 0.0f) atomicMax((int*)addr, __float_as_int(val));
  else             atomicMin((unsigned int*)addr, __float_as_uint(val));
}

// ---------- weight f32[K,M] -> f16 transposed [M,K] ----------
__global__ void cvt_f16_tr_kernel(const float* __restrict__ in, _Float16* __restrict__ out,
                                  int K, int M) {
  int t = blockIdx.x * blockDim.x + threadIdx.x;
  if (t >= K * M) return;
  int k = t / M, m = t % M;
  out[(size_t)m * K + k] = (_Float16)in[t];
}

// ---------- WMMA GEMM: Y[N,M] = X[N,K] @ W[K,M] + bias ----------
// Block: 256 thr (8 waves), 64x64 output tile. LDS-staged, ds_load_b128 fragments,
// 2 x v_wmma_f32_16x16x32_f16 per wave per K-step (A fragment reused).
// Wt is the weight matrix pre-transposed to [M][K] f16 so tiles are K-contiguous.
__global__ __launch_bounds__(256) void wmma_gemm_bias(
    const float* __restrict__ X, const _Float16* __restrict__ Wt,
    const float* __restrict__ bias, float* __restrict__ Y, int K, int M) {
  __shared__ _Float16 Xs[64][32];   // [row within block][k within step]
  __shared__ _Float16 Ws[64][32];   // [col within block][k within step]

  const int tid   = threadIdx.x;
  const int lane  = tid & 31;
  const int wave  = tid >> 5;        // 0..7
  const int rTile = wave & 3;        // 16-row tile within 64-row block
  const int cGrp  = wave >> 2;       // 0..1 -> col tiles 2*cGrp, 2*cGrp+1
  const int hsel  = lane >> 4;       // K-half group held by this lane
  const int mn    = lane & 15;       // row (A) / col (B,C) within tile
  const long long rowBase = (long long)blockIdx.x * 64;
  const int colBase = blockIdx.y * 64;

  v8f acc0 = {}, acc1 = {};
  for (int k0 = 0; k0 < K; k0 += 32) {
    __syncthreads();
    // ---- stage X tile: 64 rows x 32 k, float4 global loads -> f16 in LDS ----
#pragma unroll
    for (int i = 0; i < 2; ++i) {
      const int idx = tid + 256 * i;          // 0..511
      const int r   = idx >> 3;               // 0..63
      const int c4  = idx & 7;                // float4 chunk
      long long gr = rowBase + r; if (gr > NNODES - 1) gr = NNODES - 1;
      const float4 v = *(const float4*)(X + gr * K + k0 + 4 * c4);
      _Float16* d = &Xs[r][4 * c4];
      d[0] = (_Float16)v.x; d[1] = (_Float16)v.y;
      d[2] = (_Float16)v.z; d[3] = (_Float16)v.w;
    }
    // ---- stage W tile: 64 cols x 32 k, 8-byte copies of pre-transposed f16 ----
#pragma unroll
    for (int i = 0; i < 2; ++i) {
      const int idx = tid + 256 * i;          // 0..511
      const int c   = idx >> 3;               // 0..63
      const int c4  = idx & 7;                // half4 chunk
      int gc = colBase + c; if (gc > M - 1) gc = M - 1;
      const uint2 wv = *(const uint2*)(Wt + (size_t)gc * K + k0 + 4 * c4);
      *(uint2*)&Ws[c][4 * c4] = wv;
    }
    __syncthreads();

    // ---- fragments: two 16B LDS reads each (K-contiguous halves) ----
    const int arow = rTile * 16 + mn;
    const v8h a0 = *(const v8h*)&Xs[arow][hsel * 8];
    const v8h a1 = *(const v8h*)&Xs[arow][16 + hsel * 8];
    const v16h a = __builtin_shufflevector(a0, a1, 0,1,2,3,4,5,6,7,8,9,10,11,12,13,14,15);

    const int bc0 = (cGrp * 2 + 0) * 16 + mn;
    const v8h b00 = *(const v8h*)&Ws[bc0][hsel * 8];
    const v8h b01 = *(const v8h*)&Ws[bc0][16 + hsel * 8];
    const v16h b0 = __builtin_shufflevector(b00, b01, 0,1,2,3,4,5,6,7,8,9,10,11,12,13,14,15);

    const int bc1 = (cGrp * 2 + 1) * 16 + mn;
    const v8h b10 = *(const v8h*)&Ws[bc1][hsel * 8];
    const v8h b11 = *(const v8h*)&Ws[bc1][16 + hsel * 8];
    const v16h b1 = __builtin_shufflevector(b10, b11, 0,1,2,3,4,5,6,7,8,9,10,11,12,13,14,15);

    acc0 = __builtin_amdgcn_wmma_f32_16x16x32_f16(false, a, false, b0, (short)0, acc0, false, false);
    acc1 = __builtin_amdgcn_wmma_f32_16x16x32_f16(false, a, false, b1, (short)0, acc1, false, false);
  }

  // ---- store with bias; guard edge rows/cols ----
  const int n0 = colBase + (cGrp * 2 + 0) * 16 + mn;
  const int n1 = colBase + (cGrp * 2 + 1) * 16 + mn;
  const float bv0 = bias[n0 < M ? n0 : 0];
  const float bv1 = bias[n1 < M ? n1 : 0];
#pragma unroll
  for (int v = 0; v < 8; ++v) {
    const long long gm = rowBase + rTile * 16 + v + 8 * hsel;  // C: VGPR v -> M=v / v+8
    if (gm < NNODES) {
      if (n0 < M) Y[gm * M + n0] = acc0[v] + bv0;
      if (n1 < M) Y[gm * M + n1] = acc1[v] + bv1;
    }
  }
}

// ---------- generic fill ----------
__global__ void fill_f32_kernel(float* __restrict__ p, float v, long long n) {
  long long t = (long long)blockIdx.x * blockDim.x + threadIdx.x;
  if (t < n) p[t] = v;
}

// ---------- edge attention logits: leakyrelu(hs[src]+hd[dst]) . attn[h] ----------
__global__ void edge_logits_kernel(const float* __restrict__ HS, const float* __restrict__ HD,
                                   const float* __restrict__ attn, const int* __restrict__ src,
                                   const int* __restrict__ dst, float* __restrict__ logits, int D) {
  int t = blockIdx.x * blockDim.x + threadIdx.x;
  if (t >= NEDGES * NHEADS) return;
  int e = t >> 2, h = t & 3;
  int sn = src[e], dn = dst[e];
  const float* ps = HS + (size_t)sn * NHEADS * D + h * D;
  const float* pd = HD + (size_t)dn * NHEADS * D + h * D;
  const float* pa = attn + h * D;
  float acc = 0.0f;
  for (int d = 0; d < D; ++d) {
    float v = ps[d] + pd[d];
    v = (v > 0.0f) ? v : 0.2f * v;
    acc = fmaf(v, pa[d], acc);
  }
  logits[t] = acc;
}

// ---------- segment max over dst ----------
__global__ void edge_segmax_kernel(const float* __restrict__ logits, const int* __restrict__ dst,
                                   float* __restrict__ m) {
  int t = blockIdx.x * blockDim.x + threadIdx.x;
  if (t >= NEDGES * NHEADS) return;
  int e = t >> 2, h = t & 3;
  atomicMaxF(&m[(size_t)dst[e] * NHEADS + h], logits[t]);
}

// ---------- z = exp(logit - m[dst]); s[dst] += z ----------
__global__ void edge_expsum_kernel(float* __restrict__ logits, const int* __restrict__ dst,
                                   const float* __restrict__ m, float* __restrict__ s) {
  int t = blockIdx.x * blockDim.x + threadIdx.x;
  if (t >= NEDGES * NHEADS) return;
  int e = t >> 2, h = t & 3;
  size_t idx = (size_t)dst[e] * NHEADS + h;
  float z = __expf(logits[t] - m[idx]);
  logits[t] = z;
  atomicAdd(&s[idx], z);
}

// ---------- acc[dst] += hs[src] * (z / s[dst]) ----------
__global__ void edge_aggregate_kernel(const float* __restrict__ logits, const float* __restrict__ s,
                                      const float* __restrict__ HS, const int* __restrict__ src,
                                      const int* __restrict__ dst, float* __restrict__ acc, int D) {
  int t = blockIdx.x * blockDim.x + threadIdx.x;
  if (t >= NEDGES * NHEADS) return;
  int e = t >> 2, h = t & 3;
  int sn = src[e], dn = dst[e];
  float a = logits[t] / s[(size_t)dn * NHEADS + h];
  const float* ps = HS + (size_t)sn * NHEADS * D + h * D;
  float* pd = acc + (size_t)dn * NHEADS * D + h * D;
  __builtin_prefetch(ps, 0, 1);               // global_prefetch_b8
  for (int d = 0; d < D; ++d) atomicAdd(&pd[d], ps[d] * a);
}

// ---------- acc += res ----------
__global__ void vadd_kernel(float* __restrict__ acc, const float* __restrict__ res, long long n) {
  long long t = (long long)blockIdx.x * blockDim.x + threadIdx.x;
  if (t < n) acc[t] += res[t];
}

// ---------- mean over heads: [N, H, OUT] -> [N, OUT] ----------
__global__ void head_mean_kernel(const float* __restrict__ X, float* __restrict__ out, int OUTD) {
  int t = blockIdx.x * blockDim.x + threadIdx.x;
  if (t >= NNODES * OUTD) return;
  int n = t / OUTD, d = t % OUTD;
  const float* p = X + (size_t)n * NHEADS * OUTD + d;
  float s = 0.0f;
  for (int h = 0; h < NHEADS; ++h) s += p[h * OUTD];
  out[t] = s * 0.25f;
}

// ============================================================================
extern "C" void kernel_launch(void* const* d_in, const int* in_sizes, int n_in,
                              void* d_out, int out_size, void* d_ws, size_t ws_size,
                              hipStream_t stream) {
  (void)in_sizes; (void)n_in; (void)out_size; (void)ws_size;

  const float* x_in   = (const float*)d_in[0];
  const int*   src    = (const int*)d_in[1];
  const int*   dst    = (const int*)d_in[2];
  const float* w_src0 = (const float*)d_in[3];  const float* b_src0 = (const float*)d_in[4];
  const float* w_dst0 = (const float*)d_in[5];  const float* b_dst0 = (const float*)d_in[6];
  const float* attn0  = (const float*)d_in[7];
  const float* w_res0 = (const float*)d_in[8];  const float* b_res0 = (const float*)d_in[9];
  const float* w_src1 = (const float*)d_in[10]; const float* b_src1 = (const float*)d_in[11];
  const float* w_dst1 = (const float*)d_in[12]; const float* b_dst1 = (const float*)d_in[13];
  const float* attn1  = (const float*)d_in[14];
  const float* w_src2 = (const float*)d_in[15]; const float* b_src2 = (const float*)d_in[16];
  const float* w_dst2 = (const float*)d_in[17]; const float* b_dst2 = (const float*)d_in[18];
  const float* attn2  = (const float*)d_in[19];
  const float* w_res2 = (const float*)d_in[20]; const float* b_res2 = (const float*)d_in[21];

  // ---- workspace carve-up ----
  size_t off = 0;
  auto alloc = [&](size_t bytes) -> void* {
    void* p = (char*)d_ws + off;
    off += (bytes + 255) & ~(size_t)255;
    return p;
  };
  const size_t NF = (size_t)NNODES * 256;
  float* HS  = (float*)alloc(NF * 4);
  float* HD  = (float*)alloc(NF * 4);
  float* RES = (float*)alloc(NF * 4);
  float* P0  = (float*)alloc(NF * 4);
  float* P1  = (float*)alloc(NF * 4);
  float* LOG = (float*)alloc((size_t)NEDGES * NHEADS * 4);
  float* Mb  = (float*)alloc((size_t)NNODES * NHEADS * 4);
  float* Sb  = (float*)alloc((size_t)NNODES * NHEADS * 4);
  _Float16* whs0 = (_Float16*)alloc(128 * 256 * 2);
  _Float16* whd0 = (_Float16*)alloc(128 * 256 * 2);
  _Float16* whr0 = (_Float16*)alloc(128 * 256 * 2);
  _Float16* whs1 = (_Float16*)alloc(256 * 256 * 2);
  _Float16* whd1 = (_Float16*)alloc(256 * 256 * 2);
  _Float16* whs2 = (_Float16*)alloc(256 * 160 * 2);
  _Float16* whd2 = (_Float16*)alloc(256 * 160 * 2);
  _Float16* whr2 = (_Float16*)alloc(256 * 160 * 2);

  auto cvt = [&](const float* w, _Float16* wh, int K, int M) {
    int n = K * M;
    cvt_f16_tr_kernel<<<(n + 255) / 256, 256, 0, stream>>>(w, wh, K, M);
  };
  cvt(w_src0, whs0, 128, 256); cvt(w_dst0, whd0, 128, 256); cvt(w_res0, whr0, 128, 256);
  cvt(w_src1, whs1, 256, 256); cvt(w_dst1, whd1, 256, 256);
  cvt(w_src2, whs2, 256, 160); cvt(w_dst2, whd2, 256, 160); cvt(w_res2, whr2, 256, 160);

  auto gemm = [&](const float* X, const _Float16* Wt, const float* bias, float* Y, int K, int M) {
    dim3 grid((NNODES + 63) / 64, (M + 63) / 64);
    wmma_gemm_bias<<<grid, 256, 0, stream>>>(X, Wt, bias, Y, K, M);
  };
  auto fill = [&](float* p, float v, long long n) {
    fill_f32_kernel<<<(unsigned)((n + 255) / 256), 256, 0, stream>>>(p, v, n);
  };
  const int EH_BLK = (NEDGES * NHEADS + 255) / 256;
  auto edge_phase = [&](const float* attn, float* acc, int D) {
    fill(Mb, -__builtin_inff(), (long long)NNODES * NHEADS);
    fill(Sb, 0.0f, (long long)NNODES * NHEADS);
    fill(acc, 0.0f, (long long)NNODES * NHEADS * D);
    edge_logits_kernel<<<EH_BLK, 256, 0, stream>>>(HS, HD, attn, src, dst, LOG, D);
    edge_segmax_kernel<<<EH_BLK, 256, 0, stream>>>(LOG, dst, Mb);
    edge_expsum_kernel<<<EH_BLK, 256, 0, stream>>>(LOG, dst, Mb, Sb);
    edge_aggregate_kernel<<<EH_BLK, 256, 0, stream>>>(LOG, Sb, HS, src, dst, acc, D);
  };
  auto vadd = [&](float* acc, const float* res, long long n) {
    vadd_kernel<<<(unsigned)((n + 255) / 256), 256, 0, stream>>>(acc, res, n);
  };

  // ---- layer 0: in=128, per-head out=64, linear residual ----
  gemm(x_in, whs0, b_src0, HS, 128, 256);
  gemm(x_in, whd0, b_dst0, HD, 128, 256);
  gemm(x_in, whr0, b_res0, RES, 128, 256);
  edge_phase(attn0, P0, 64);
  vadd(P0, RES, (long long)NNODES * 256);

  // ---- layer 1: in=256 == H*HID, identity residual ----
  gemm(P0, whs1, b_src1, HS, 256, 256);
  gemm(P0, whd1, b_dst1, HD, 256, 256);
  edge_phase(attn1, P1, 64);
  vadd(P1, P0, (long long)NNODES * 256);

  // ---- layer 2: in=256, per-head out=40, linear residual ----
  gemm(P1, whs2, b_src2, HS, 256, 160);
  gemm(P1, whd2, b_dst2, HD, 256, 160);
  gemm(P1, whr2, b_res2, RES, 256, 160);
  edge_phase(attn2, P0, 40);
  vadd(P0, RES, (long long)NNODES * 160);

  // ---- mean over heads ----
  head_mean_kernel<<<(NNODES * 40 + 255) / 256, 256, 0, stream>>>(P0, (float*)d_out, 40);
}
